// GATConv_30494267801712
// MI455X (gfx1250) — matrix operand order repaired and checked by reading, not compiled, not run on hardware
//
#include <hip/hip_runtime.h>
#include <hip/hip_bf16.h>

// ---------------------------------------------------------------------------
// GAT layer for MI455X (gfx1250, wave32).
//
// Roofline: GEMM h=x@W is 13.1 GFLOP over ~154MB -> memory-bound (~6.6us at
// 23.3 TB/s), so we use bf16 WMMA (v_wmma_f32_16x16x32_bf16, f32 accum).
// W in bf16 is 512*256*2 = 256KB -> fits in the 320KB WGP LDS, so each block
// stages ALL of W once (pre-swizzled into B-fragment lane order) and then
// streams M-tiles. Edge phase: h (51MB) and out (51MB) both fit in the 192MB
// L2, so the 800k-edge gather + atomic scatter runs at L2 bandwidth.
// ---------------------------------------------------------------------------

#define N_NODES 50000
#define N_EDGES 800000
#define IN_DIM  512
#define HEADS   4
#define OUT_DIM 64
#define N_FEAT  (HEADS * OUT_DIM)   // 256
#define M_TILES (N_NODES / 16)      // 3125 (exact)
#define K_TILES (IN_DIM / 32)       // 16
#define N_TILES (N_FEAT / 16)       // 16

typedef __attribute__((ext_vector_type(16))) __bf16 v16bf;
typedef __attribute__((ext_vector_type(8)))  float  v8f;

// ---------------------------------------------------------------------------
// Kernel 1: h = x @ W via WMMA bf16. One wave computes a 16(M) x 256(N) strip
// (16 accumulators). W lives fully in LDS, pre-swizzled so a lane's B-fragment
// is 32 contiguous bytes (two ds_load_b128).
//
// B-frag layout (bf16 32x16): lanes 0..15 hold K=0..15 (2 halves / VGPR),
// lanes 16..31 hold K=16..31; N = lane & 15.
// A-frag layout (bf16 16x32): lane L holds row M = L&15; K = kb..kb+15 with
// kb = 16*(L>=16).
// ---------------------------------------------------------------------------
__global__ __launch_bounds__(256, 1)
void gat_gemm_wmma(const float* __restrict__ x, const float* __restrict__ W,
                   float* __restrict__ hbuf) {
    extern __shared__ __bf16 ldsW[];   // 512*256 bf16 = 256KB (fits 320KB WGP LDS)
    const int tid = threadIdx.x;

    // Stage + swizzle W: element (k,n) -> fragment(kt,nt), lane, half j.
    for (int idx = tid; idx < IN_DIM * N_FEAT; idx += 256) {
        const int k  = idx >> 8;          // W is [K][N] row-major, N=256
        const int n  = idx & 255;
        const int kt = k >> 5, kk = k & 31;
        const int nt = n >> 4, nn = n & 15;
        const int ln = nn + (kk & 16);    // +16 lane half if kk >= 16
        const int j  = kk & 15;
        ldsW[((((kt * N_TILES) + nt) << 5) + ln) * 16 + j] = (__bf16)W[idx];
    }
    __syncthreads();

    const int wave = tid >> 5;
    const int lane = tid & 31;
    const int nlo  = lane & 15;     // row (A) / column (B,C)
    const int khi  = lane >> 4;     // lane half

    for (int tile = blockIdx.x * 8 + wave; tile < M_TILES; tile += gridDim.x * 8) {
        const int mBase = tile << 4;
        v8f c[N_TILES] = {};        // 16 accumulators = 128 VGPRs

        const float* xrow = x + (size_t)(mBase + nlo) * IN_DIM + (khi << 4);

        #pragma unroll
        for (int kt = 0; kt < K_TILES; ++kt) {
            const float* xp = xrow + kt * 32;
            const float4 f0 = *(const float4*)(xp +  0);
            const float4 f1 = *(const float4*)(xp +  4);
            const float4 f2 = *(const float4*)(xp +  8);
            const float4 f3 = *(const float4*)(xp + 12);
            v16bf a;
            a[0]  = (__bf16)f0.x; a[1]  = (__bf16)f0.y; a[2]  = (__bf16)f0.z; a[3]  = (__bf16)f0.w;
            a[4]  = (__bf16)f1.x; a[5]  = (__bf16)f1.y; a[6]  = (__bf16)f1.z; a[7]  = (__bf16)f1.w;
            a[8]  = (__bf16)f2.x; a[9]  = (__bf16)f2.y; a[10] = (__bf16)f2.z; a[11] = (__bf16)f2.w;
            a[12] = (__bf16)f3.x; a[13] = (__bf16)f3.y; a[14] = (__bf16)f3.z; a[15] = (__bf16)f3.w;

            #pragma unroll
            for (int nt = 0; nt < N_TILES; ++nt) {
                const v16bf b =
                    *(const v16bf*)&ldsW[((((kt * N_TILES) + nt) << 5) + lane) * 16];
                c[nt] = __builtin_amdgcn_wmma_f32_16x16x32_bf16(
                    /*neg_a=*/false, a, /*neg_b=*/false, b,
                    /*c_mod=*/(short)0, c[nt], /*reuse_a=*/false, /*reuse_b=*/false);
            }
        }

        // C/D layout: lane half khi -> rows r + 8*khi, column nt*16 + nlo.
        #pragma unroll
        for (int nt = 0; nt < N_TILES; ++nt) {
            float* op = hbuf + (size_t)(mBase + (khi << 3)) * N_FEAT + nt * 16 + nlo;
            #pragma unroll
            for (int r = 0; r < 8; ++r) op[(size_t)r * N_FEAT] = c[nt][r];
        }
    }
}

// ---------------------------------------------------------------------------
// Ordered-uint encoding for float atomicMax (monotone map IEEE f32 -> u32).
// ---------------------------------------------------------------------------
__device__ __forceinline__ unsigned f2ord(float f) {
    unsigned u = __float_as_uint(f);
    return (u & 0x80000000u) ? ~u : (u | 0x80000000u);
}
__device__ __forceinline__ float ord2f(unsigned u) {
    u = (u & 0x80000000u) ? (u & 0x7fffffffu) : ~u;
    return __uint_as_float(u);
}

// Kernel 2: out = bias (broadcast); init segment max (ordered -inf) and sums.
__global__ void gat_init(const float* __restrict__ bias, float* __restrict__ out,
                         unsigned* __restrict__ maxe, float* __restrict__ sume) {
    const long long i = (long long)blockIdx.x * blockDim.x + threadIdx.x;
    if (i < (long long)N_NODES * N_FEAT) out[i] = bias[i & 255];
    if (i < (long long)N_NODES * HEADS) { maxe[i] = 0u; sume[i] = 0.0f; }
}

// Kernel 3: per-(node,head) attention scores s_src/s_dst (64-d dot products).
__global__ void gat_scores(const float* __restrict__ hbuf, const float* __restrict__ a,
                           float* __restrict__ s_src, float* __restrict__ s_dst) {
    const int t = blockIdx.x * blockDim.x + threadIdx.x;
    if (t >= N_NODES * HEADS) return;
    const int node = t >> 2, head = t & 3;
    const float4* hp = (const float4*)(hbuf + (size_t)node * N_FEAT + head * OUT_DIM);
    const float4* as = (const float4*)(a + head * 2 * OUT_DIM);
    const float4* ad = (const float4*)(a + head * 2 * OUT_DIM + OUT_DIM);
    float ss = 0.0f, sd = 0.0f;
    #pragma unroll
    for (int i = 0; i < OUT_DIM / 4; ++i) {
        const float4 hv = hp[i], av = as[i], dv = ad[i];
        ss = fmaf(hv.x, av.x, fmaf(hv.y, av.y, fmaf(hv.z, av.z, fmaf(hv.w, av.w, ss))));
        sd = fmaf(hv.x, dv.x, fmaf(hv.y, dv.y, fmaf(hv.z, dv.z, fmaf(hv.w, dv.w, sd))));
    }
    s_src[t] = ss; s_dst[t] = sd;
}

// Kernel 4: e = leaky_relu(s_src[src]+s_dst[dst]); segment max over src.
__global__ void gat_edge_logits(const int* __restrict__ edges,
                                const float* __restrict__ s_src,
                                const float* __restrict__ s_dst,
                                float* __restrict__ ebuf, unsigned* __restrict__ maxe) {
    const int t = blockIdx.x * blockDim.x + threadIdx.x;
    if (t >= N_EDGES * HEADS) return;
    const int e = t >> 2, hd = t & 3;
    const int s = edges[2 * e], d = edges[2 * e + 1];
    float v = s_src[s * HEADS + hd] + s_dst[d * HEADS + hd];
    v = v > 0.0f ? v : 0.2f * v;
    ebuf[t] = v;
    atomicMax(&maxe[s * HEADS + hd], f2ord(v));
}

// Kernel 5: exp(e - max) and segment sum over src.
__global__ void gat_edge_exp(const int* __restrict__ edges, float* __restrict__ ebuf,
                             const unsigned* __restrict__ maxe, float* __restrict__ sume) {
    const int t = blockIdx.x * blockDim.x + threadIdx.x;
    if (t >= N_EDGES * HEADS) return;
    const int e = t >> 2, hd = t & 3;
    const int s = edges[2 * e];
    const float ex = __expf(ebuf[t] - ord2f(maxe[s * HEADS + hd]));
    ebuf[t] = ex;
    atomicAdd(&sume[s * HEADS + hd], ex);
}

// Kernel 6: alpha = exp_e / (sum_exp[src] + 1e-10), in place.
__global__ void gat_edge_alpha(const int* __restrict__ edges, float* __restrict__ ebuf,
                               const float* __restrict__ sume) {
    const int t = blockIdx.x * blockDim.x + threadIdx.x;
    if (t >= N_EDGES * HEADS) return;
    const int e = t >> 2, hd = t & 3;
    const int s = edges[2 * e];
    ebuf[t] = ebuf[t] / (sume[s * HEADS + hd] + 1e-10f);
}

// Kernel 7: out[dst] += alpha * h[src]. One block per edge, one thread per
// channel: coalesced 256-float gather from h (L2-resident) + f32 atomic scatter.
__global__ void gat_scatter(const int* __restrict__ edges, const float* __restrict__ ebuf,
                            const float* __restrict__ hbuf, float* __restrict__ out) {
    const int e = blockIdx.x;
    const int c = threadIdx.x;
    const int s = edges[2 * e], d = edges[2 * e + 1];
    const int hd = c >> 6;
    const float alpha = ebuf[e * HEADS + hd];
    atomicAdd(&out[(size_t)d * N_FEAT + c], alpha * hbuf[(size_t)s * N_FEAT + c]);
}

// ---------------------------------------------------------------------------
extern "C" void kernel_launch(void* const* d_in, const int* in_sizes, int n_in,
                              void* d_out, int out_size, void* d_ws, size_t ws_size,
                              hipStream_t stream) {
    const float* x     = (const float*)d_in[0];   // 50000 x 512
    const int*   edges = (const int*)  d_in[1];   // 800000 x 2
    const float* W     = (const float*)d_in[2];   // 512 x 256
    const float* a     = (const float*)d_in[3];   // 4 x 128
    const float* bias  = (const float*)d_in[4];   // 256
    float* out = (float*)d_out;                   // 50000 x 256

    // Workspace layout (~67.2 MB total).
    float*    hbuf  = (float*)d_ws;                                   // 12.8M f32
    float*    s_src = hbuf + (size_t)N_NODES * N_FEAT;                // 200k
    float*    s_dst = s_src + (size_t)N_NODES * HEADS;                // 200k
    float*    sume  = s_dst + (size_t)N_NODES * HEADS;                // 200k
    unsigned* maxe  = (unsigned*)(sume + (size_t)N_NODES * HEADS);    // 200k
    float*    ebuf  = (float*)(maxe + (size_t)N_NODES * HEADS);       // 3.2M

    const size_t ldsBytes = (size_t)IN_DIM * N_FEAT * sizeof(__bf16); // 256KB

    gat_gemm_wmma<<<dim3(128), dim3(256), ldsBytes, stream>>>(x, W, hbuf);

    const long long initN = (long long)N_NODES * N_FEAT;
    gat_init<<<dim3((unsigned)((initN + 255) / 256)), dim3(256), 0, stream>>>(
        bias, out, maxe, sume);

    gat_scores<<<dim3((N_NODES * HEADS + 255) / 256), dim3(256), 0, stream>>>(
        hbuf, a, s_src, s_dst);

    const int eh = N_EDGES * HEADS;
    gat_edge_logits<<<dim3((eh + 255) / 256), dim3(256), 0, stream>>>(
        edges, s_src, s_dst, ebuf, maxe);
    gat_edge_exp<<<dim3((eh + 255) / 256), dim3(256), 0, stream>>>(
        edges, ebuf, maxe, sume);
    gat_edge_alpha<<<dim3((eh + 255) / 256), dim3(256), 0, stream>>>(
        edges, ebuf, sume);

    gat_scatter<<<dim3(N_EDGES), dim3(256), 0, stream>>>(edges, ebuf, hbuf, out);
}
